// CorrectedContrastiveLoss_19421842112637
// MI455X (gfx1250) — matrix operand order repaired and checked by reading, not compiled, not run on hardware
//
#include <hip/hip_runtime.h>
#include <hip/hip_bf16.h>

typedef __attribute__((ext_vector_type(16))) _Float16 v16h;
typedef __attribute__((ext_vector_type(8)))  _Float16 v8h;
typedef __attribute__((ext_vector_type(4)))  _Float16 v4h;
typedef __attribute__((ext_vector_type(8)))  float    v8f;
typedef __attribute__((ext_vector_type(4)))  float    v4f;

// LDS (address-space 3) flavors: 32-bit pointers, ds_* instructions.
typedef __attribute__((address_space(3))) char as3_char;
typedef __attribute__((address_space(3))) const v8h as3_v8h;

#define D        128
#define B_ROWS   4096
#define N_ROWS   8192
#define NTILES   512          // 8192 / 16 column tiles
#define CSPLIT   8            // column splits for parallelism (4096 waves total)
#define TPS      (NTILES / CSPLIT)
#define LDS_STRIDE 272        // 128 halves * 2B + 16B pad (odd multiple of 16B -> conflict-light)
#define BUFBYTES (16 * LDS_STRIDE)

union FragH { v16h v; v8h h[2]; };

// ---------------------------------------------------------------- zero scratch
__global__ void zero_rowacc(float* __restrict__ rowacc) {
  int i = blockIdx.x * blockDim.x + threadIdx.x;
  if (i < N_ROWS) rowacc[i] = 0.0f;
}

// ------------------------------------------------- row L2-normalize -> f16 reps
__global__ __launch_bounds__(256) void normalize_rows(const float* __restrict__ zi,
                                                      const float* __restrict__ zj,
                                                      _Float16* __restrict__ reps) {
  const int row  = blockIdx.x * 8 + (threadIdx.x >> 5);  // one wave per row
  const int lane = threadIdx.x & 31;
  const float* src = (row < B_ROWS) ? (zi + (size_t)row * D)
                                    : (zj + (size_t)(row - B_ROWS) * D);
  v4f v = *(const v4f*)(src + lane * 4);
  float ss = v.x * v.x + v.y * v.y + v.z * v.z + v.w * v.w;
  ss += __shfl_xor(ss, 1, 32);
  ss += __shfl_xor(ss, 2, 32);
  ss += __shfl_xor(ss, 4, 32);
  ss += __shfl_xor(ss, 8, 32);
  ss += __shfl_xor(ss, 16, 32);
  const float rn = rsqrtf(ss);
  v4h h;
  h.x = (_Float16)(v.x * rn);
  h.y = (_Float16)(v.y * rn);
  h.z = (_Float16)(v.z * rn);
  h.w = (_Float16)(v.w * rn);
  *(v4h*)(reps + (size_t)row * D + lane * 4) = h;
}

// Async memory->LDS staging of one 16-row reps tile (32B per thread, 2 x b128).
// The offset:16 immediate is added to both the global and the LDS address.
__device__ __forceinline__ void stage_tile_async(const _Float16* __restrict__ reps,
                                                 int tile, int srow, int sbyte) {
  const unsigned ldsOff = (unsigned)((tile & 1) * BUFBYTES + srow * LDS_STRIDE + sbyte);
  const char* g = (const char*)(reps + (size_t)(tile * 16 + srow) * D) + sbyte;
  asm volatile("global_load_async_to_lds_b128 %0, %1, off\n\t"
               "global_load_async_to_lds_b128 %0, %1, off offset:16"
               :: "v"(ldsOff), "v"(g) : "memory");
}

// ------------------------------------------------------- fused WMMA + online LSE
__global__ __launch_bounds__(128) void simclr_tiles(const _Float16* __restrict__ reps,
                                                    float* __restrict__ rowacc,
                                                    float* __restrict__ rowpos) {
  __shared__ __align__(16) char lds[2 * BUFBYTES];

  // LDS is written only by inline-asm async loads. Launder the base pointer
  // (as an AS3 pointer, so reads stay ds_load_b128) to stop the compiler from
  // folding loads of a "never stored" object to undef.
  as3_char* ldsbase = (as3_char*)lds;
  asm volatile("" : "+v"(ldsbase));

  const int tid    = threadIdx.x;
  const int lane   = tid & 31;
  const int wave   = tid >> 5;
  const int halfId = lane >> 4;   // 0: K-low / M 0-7 ; 1: K-high / M 8-15
  const int ln     = lane & 15;   // A: row-in-tile, B/C: column-in-tile
  const int i0     = (blockIdx.x * 4 + wave) << 4;

  // Resident A fragments (rows i0..i0+15, K=0..127), CDNA5 16-bit A layout:
  // lane<16 holds K {kk..kk+7, kk+16..kk+23}; lane>=16 holds {kk+8.., kk+24..}.
  FragH a[4];
  const _Float16* arow = reps + (size_t)(i0 + ln) * D;
#pragma unroll
  for (int c = 0; c < 4; ++c) {
    const int kk = c * 32;
    a[c].h[0] = *(const v8h*)(arow + kk + halfId * 8);
    a[c].h[1] = *(const v8h*)(arow + kk + 16 + halfId * 8);
  }

  float s[8], pos[8], dsub[8];
#pragma unroll
  for (int r = 0; r < 8; ++r) { s[r] = 0.0f; pos[r] = 0.0f; dsub[r] = 0.0f; }

  const int diagTile = i0 >> 4;
  const int posTile  = ((i0 + B_ROWS) & (N_ROWS - 1)) >> 4;
  const int tbeg = blockIdx.y * TPS;
  const int tend = tbeg + TPS;

  // LDS staging: 128 threads move 16 rows x 256B (32B per thread), coalesced.
  const int srow  = tid >> 3;
  const int sbyte = (tid & 7) * 32;

  stage_tile_async(reps, tbeg, srow, sbyte);           // prologue
  asm volatile("s_wait_asynccnt 0x0" ::: "memory");
  __syncthreads();

  const float KX = 2.8853900817779268f;  // 2 * log2(e): sim/T folded into exp2

  for (int t = tbeg; t < tend; ++t) {
    if (t + 1 < tend)                                  // prefetch next tile -> LDS
      stage_tile_async(reps, t + 1, srow, sbyte);

    // Preload ALL B fragments first (distinct values -> distinct registers ->
    // one ds_load clause + partial dscnt waits pipelined under the WMMAs).
    const as3_char* buf = ldsbase + (t & 1) * BUFBYTES + ln * LDS_STRIDE;
    FragH b[4];
#pragma unroll
    for (int c = 0; c < 4; ++c) {
      const as3_char* p = buf + c * 64 + halfId * 16;
      b[c].h[0] = *(as3_v8h*)p;
      b[c].h[1] = *(as3_v8h*)(p + 32);
    }
    v8f acc = {0.f, 0.f, 0.f, 0.f, 0.f, 0.f, 0.f, 0.f};
#pragma unroll
    for (int c = 0; c < 4; ++c)
      acc = __builtin_amdgcn_wmma_f32_16x16x32_f16(false, a[c].v, false, b[c].v,
                                                   (short)0, acc, false, false);

    // Branch-free exp-sum (sim in [-2,2] -> raw v_exp_f32 is safe).
    // C layout: VGPR r, lane -> row M = r + 8*halfId, col N = ln.
    float e[8];
#pragma unroll
    for (int r = 0; r < 8; ++r) {
      e[r] = __builtin_amdgcn_exp2f(acc[r] * KX);
      s[r] += e[r];
    }
    // Diagonal handled by capture-and-subtract after the loop (identical e).
    if (t == diagTile) {
#pragma unroll
      for (int r = 0; r < 8; ++r)
        dsub[r] = (ln == r + 8 * halfId) ? e[r] : dsub[r];
    }
    if (t == posTile) {
#pragma unroll
      for (int r = 0; r < 8; ++r)
        pos[r] = (ln == r + 8 * halfId) ? acc[r] * 2.0f : pos[r];
    }

    asm volatile("s_wait_asynccnt 0x0" ::: "memory");  // next buffer fully in LDS
    __syncthreads();
  }

  // Remove the self-similarity term, reduce 16 columns per half-wave, publish.
  const bool ownsPos = (posTile >= tbeg) && (posTile < tend);
#pragma unroll
  for (int r = 0; r < 8; ++r) {
    float rs = s[r] - dsub[r];
    rs += __shfl_xor(rs, 1, 32);
    rs += __shfl_xor(rs, 2, 32);
    rs += __shfl_xor(rs, 4, 32);
    rs += __shfl_xor(rs, 8, 32);
    if (ln == r + 8 * halfId) {           // exactly one lane per row
      const int row = i0 + r + 8 * halfId;
      atomicAdd(&rowacc[row], rs);
      if (ownsPos) rowpos[row] = pos[r];
    }
  }
}

// --------------------------------------------------------------- final mean
__global__ __launch_bounds__(256) void finalize_loss(const float* __restrict__ rowacc,
                                                     const float* __restrict__ rowpos,
                                                     float* __restrict__ out) {
  __shared__ float red[256];
  const float LN2 = 0.6931471805599453f;
  float t = 0.0f;
  for (int i = threadIdx.x; i < N_ROWS; i += 256)
    t += __builtin_amdgcn_logf(rowacc[i]) * LN2 - rowpos[i];  // ln(sum) - pos
  red[threadIdx.x] = t;
  __syncthreads();
  for (int sft = 128; sft > 0; sft >>= 1) {
    if (threadIdx.x < sft) red[threadIdx.x] += red[threadIdx.x + sft];
    __syncthreads();
  }
  if (threadIdx.x == 0) *out = red[0] * (1.0f / (float)N_ROWS);
}

// ------------------------------------------------------------------ launcher
extern "C" void kernel_launch(void* const* d_in, const int* in_sizes, int n_in,
                              void* d_out, int out_size, void* d_ws, size_t ws_size,
                              hipStream_t stream) {
  const float* zi = (const float*)d_in[0];
  const float* zj = (const float*)d_in[1];
  float* out = (float*)d_out;

  _Float16* reps = (_Float16*)d_ws;                                   // 2 MB
  float* rowacc  = (float*)((char*)d_ws + (size_t)N_ROWS * D * sizeof(_Float16));
  float* rowpos  = rowacc + N_ROWS;

  zero_rowacc<<<N_ROWS / 256, 256, 0, stream>>>(rowacc);
  normalize_rows<<<N_ROWS / 8, 256, 0, stream>>>(zi, zj, reps);
  dim3 grid(N_ROWS / 64, CSPLIT);   // 128 row-blocks x 8 column splits
  simclr_tiles<<<grid, 128, 0, stream>>>(reps, rowacc, rowpos);
  finalize_loss<<<1, 256, 0, stream>>>(rowacc, rowpos, out);
}
